// Anomalous_detection_63806034149938
// MI455X (gfx1250) — compile-verified, matrix operand order checked by reading, not run on hardware
//
#include <hip/hip_runtime.h>
#include <hip/hip_bf16.h>
#include <math.h>

// Problem constants (match reference)
#define BB   32
#define TT   64
#define LL   512
#define CC   32
#define FF   32
#define LO   256   // L/2
#define UU   32
#define SS   256   // sequence length into the LSTMs
#define MROWS (BB * LO)   // 8192 rows per ConvLSTM step GEMM

typedef __attribute__((ext_vector_type(16))) _Float16 v16h;
typedef __attribute__((ext_vector_type(8)))  float    v8f;

// CDNA5 16-bit A(16x32)/B(32x16) fragment K-index for half slot j of a lane:
// lanes 0-15 : j0..7 -> K 0..7,  j8..15 -> K 16..23
// lanes 16-31: j0..7 -> K 8..15, j8..15 -> K 24..31
__device__ __forceinline__ int klocal(int j, int lane) {
    return ((j & 8) << 1) + ((lane & 16) ? 8 : 0) + (j & 7);
}
// Inverse: given K-within-32 (kl) and tile row, which (lane, j) slot holds it
__device__ __forceinline__ int slot_lane(int kl, int row) { return (((kl >> 3) & 1) << 4) + row; }
__device__ __forceinline__ int slot_j(int kl)             { return (((kl >> 4) & 1) << 3) + (kl & 7); }

__device__ __forceinline__ float hsig(float x) {
    float v = 0.2f * x + 0.5f;
    return fminf(fmaxf(v, 0.0f), 1.0f);
}
__device__ __forceinline__ float sigm(float x) {
    return 1.0f / (1.0f + __expf(-x));
}

// ---------------------------------------------------------------------------
// Setup: emit f16 weights directly in WMMA B-fragment order and zero states.
//   WcS: [nt(8)][ks(4)][lane(32)][j(16)]  from [k_conv(64x128); r_conv(64x128)]
//   WfS/WbS: [nt(8)][ks(2)][lane(32)][j(16)] from [k(32x128); r(32x128)]
// A lane's fragment is 16 contiguous halfs (32 B) -> 2 x b128 loads.
// ---------------------------------------------------------------------------
__global__ void setup_kernel(const float* __restrict__ kconv, const float* __restrict__ rconv,
                             const float* __restrict__ kf,    const float* __restrict__ rf,
                             const float* __restrict__ kb,    const float* __restrict__ rb,
                             _Float16* __restrict__ WcS, _Float16* __restrict__ WfS,
                             _Float16* __restrict__ WbS, float* __restrict__ zero_region,
                             unsigned zero_n) {
    unsigned tid    = blockIdx.x * blockDim.x + threadIdx.x;
    unsigned stride = gridDim.x * blockDim.x;
    for (unsigned i = tid; i < 16384u; i += stride) {
        const int j = i & 15, lane = (i >> 4) & 31, ks = (i >> 9) & 3, nt = i >> 11;
        const int K = ks * 32 + klocal(j, lane);
        const int n = nt * 16 + (lane & 15);
        const float v = (K < 64) ? kconv[K * 128 + n] : rconv[(K - 64) * 128 + n];
        WcS[i] = (_Float16)v;
    }
    for (unsigned i = tid; i < 8192u; i += stride) {
        const int j = i & 15, lane = (i >> 4) & 31, ks = (i >> 9) & 1, nt = i >> 10;
        const int K = ks * 32 + klocal(j, lane);
        const int n = nt * 16 + (lane & 15);
        WfS[i] = (_Float16)((K < 32) ? kf[K * 128 + n] : rf[(K - 32) * 128 + n]);
        WbS[i] = (_Float16)((K < 32) ? kb[K * 128 + n] : rb[(K - 32) * 128 + n]);
    }
    for (unsigned i = tid; i < zero_n; i += stride) zero_region[i] = 0.0f;
}

// ---------------------------------------------------------------------------
// Swizzle the ConvLSTM hidden sequence into A-fragment-order f16 for the LSTMs:
//   hseqS: [s(256)][mtile(2)][lane(32)][j(16)]
// ---------------------------------------------------------------------------
__global__ void swz_hseq_kernel(const float* __restrict__ hseq, _Float16* __restrict__ hseqS) {
    const unsigned i = blockIdx.x * blockDim.x + threadIdx.x;   // 262144 total
    const int j = i & 15, lane = (i >> 4) & 31, mt = (i >> 9) & 1;
    const int s = i >> 10;
    const int u = klocal(j, lane);             // hidden unit 0..31 (K index)
    const int brow = mt * 16 + (lane & 15);    // batch row
    hseqS[i] = (_Float16)hseq[((size_t)brow * SS + s) * FF + u];
}

// ---------------------------------------------------------------------------
// One ConvLSTM1D timestep, input conv fused:
//   z[m,:] = [ x(b,t,2o..2o+1,:) | h_prev(b,o,:) | h_prev(b,o+1,:) ] @ Wc + b
// grid: 512 blocks (16 rows each), 256 threads = 8 waves (one N-tile each).
// Cooperative stage: coalesced f32 global reads, f16 fragment-order LDS writes.
// ---------------------------------------------------------------------------
__global__ void __launch_bounds__(256)
convlstm_step_kernel(const float* __restrict__ x, const _Float16* __restrict__ WcS,
                     const float* __restrict__ bconv,
                     const float* __restrict__ h_prev, float* __restrict__ h_next,
                     float* __restrict__ c_buf, int t) {
    __shared__ __align__(32) _Float16 a_swz[4][32][16];  // A fragments, 4 K-steps
    __shared__ float zt[16][132];                        // gate pre-activations

    const int tid  = threadIdx.x;
    const int lane = tid & 31;
    const int wave = tid >> 5;         // = N tile index (n0 = wave*16)
    const int m0   = blockIdx.x * 16;

    // ---- cooperative load: coalesced f32 reads -> swizzled f16 LDS ----
    for (int idx = tid; idx < 16 * 128; idx += 256) {
        const int rr = idx >> 7;
        const int kk = idx & 127;
        const int mm = m0 + rr;
        const int bb = mm >> 8;        // batch
        const int oo = mm & 255;       // output spatial position
        float v;
        if (kk < 64) {
            // two stride-2 input taps x(b,t,2o..2o+1,:) = 64 contiguous floats
            v = x[((size_t)(bb * TT + t) * LL + (size_t)oo * 2) * CC + kk];
        } else {
            const int j2 = kk - 64;
            if (j2 < 32) v = h_prev[(size_t)mm * FF + j2];
            else         v = (oo < 255) ? h_prev[(size_t)(mm + 1) * FF + (j2 - 32)] : 0.0f;
        }
        const int ks = kk >> 5, kl = kk & 31;
        a_swz[ks][slot_lane(kl, rr)][slot_j(kl)] = (_Float16)v;
    }
    __syncthreads();

    // ---- 4 x WMMA f16 (K = 128); A from LDS frags, B from pre-swizzled global ----
    const _Float16* wbase = WcS + (size_t)wave * 4 * 512 + (size_t)lane * 16;
    v8f acc = {};
#pragma unroll
    for (int ks = 0; ks < 4; ++ks) {
        const v16h a  = *(const v16h*)&a_swz[ks][lane][0];
        const v16h bm = *(const v16h*)(wbase + (size_t)ks * 512);
        acc = __builtin_amdgcn_wmma_f32_16x16x32_f16(false, a, false, bm,
                                                     (short)0, acc, false, false);
    }

    // ---- scatter C tile (+bias) to LDS ----
    const int ccol  = wave * 16 + (lane & 15);
    const int rbase = (lane & 16) ? 8 : 0;
    const float bc  = bconv[ccol];
#pragma unroll
    for (int r = 0; r < 8; ++r) zt[rbase + r][ccol] = acc[r] + bc;
    __syncthreads();

    // ---- fused gates: c = f*c + i*tanh(g); h = o*tanh(c) ----
    for (int idx = tid; idx < 16 * 32; idx += 256) {
        const int rr = idx >> 5;
        const int u  = idx & 31;
        const int mm = m0 + rr;
        const float zi = zt[rr][u];
        const float zf = zt[rr][u + 32];
        const float zg = zt[rr][u + 64];
        const float zo = zt[rr][u + 96];
        const float ig = hsig(zi), fg = hsig(zf), og = hsig(zo);
        const float cn = fg * c_buf[(size_t)mm * FF + u] + ig * tanhf(zg);
        c_buf[(size_t)mm * FF + u]  = cn;
        h_next[(size_t)mm * FF + u] = og * tanhf(cn);
    }
}

// ---------------------------------------------------------------------------
// Persistent bidirectional LSTM + final dense. One workgroup, 1024 thr = 32 waves:
//   dir = wave/16; within dir: 2 M-tiles (batch halves) x 8 N-tiles.
// Per step per wave: 2 x 32B fragment loads -> 2 WMMAs -> LDS -> gates.
// B fragments preloaded once; h state mirrored in fragment-order f16 LDS.
// ---------------------------------------------------------------------------
__global__ void __launch_bounds__(1024)
lstm_bidir_kernel(const _Float16* __restrict__ hseqS,
                  const _Float16* __restrict__ WfS, const _Float16* __restrict__ WbS,
                  const float* __restrict__ bf, const float* __restrict__ bbias,
                  const float* __restrict__ wd, const float* __restrict__ bd,
                  float* __restrict__ out) {
    __shared__ float hst[2][32][32];
    __shared__ float cst[2][32][32];
    __shared__ float zl[2][32][132];
    __shared__ __align__(32) _Float16 hsw[2][2][32][16];  // [dir][mtile][lane][j]

    const int tid  = threadIdx.x;
    const int lane = tid & 31;
    const int wave = tid >> 5;
    const int dir  = wave >> 4;          // 0 = forward, 1 = backward
    const int wd_  = wave & 15;
    const int mt   = wd_ >> 3;           // batch tile (0: b0-15, 1: b16-31)
    const int nt   = wd_ & 7;            // gate-column tile
    const int m0   = mt * 16;
    const int n0   = nt * 16;

    const _Float16* WS  = dir ? WbS : WfS;
    const float*    bia = dir ? bbias : bf;

    // B fragments: constant across the whole scan (one 32B load each)
    const v16h bfrag0 = *(const v16h*)(WS + ((size_t)nt * 2 + 0) * 512 + (size_t)lane * 16);
    const v16h bfrag1 = *(const v16h*)(WS + ((size_t)nt * 2 + 1) * 512 + (size_t)lane * 16);

    for (int idx = tid; idx < 2 * 32 * 32; idx += 1024) {
        (&hst[0][0][0])[idx] = 0.0f;
        (&cst[0][0][0])[idx] = 0.0f;
    }
    for (int idx = tid; idx < 2 * 2 * 32 * 16; idx += 1024)
        (&hsw[0][0][0][0])[idx] = (_Float16)0.0f;
    __syncthreads();

    const int ccol  = n0 + (lane & 15);
    const int rbase = (lane & 16) ? 8 : 0;
    const float bc  = bia[ccol];

    for (int s = 0; s < SS; ++s) {
        const int seff = dir ? (SS - 1 - s) : s;

        const v16h a0 = *(const v16h*)(hseqS + ((size_t)seff * 2 + mt) * 512 + (size_t)lane * 16);
        const v16h a1 = *(const v16h*)&hsw[dir][mt][lane][0];

        v8f acc = {};
        acc = __builtin_amdgcn_wmma_f32_16x16x32_f16(false, a0, false, bfrag0,
                                                     (short)0, acc, false, false);
        acc = __builtin_amdgcn_wmma_f32_16x16x32_f16(false, a1, false, bfrag1,
                                                     (short)0, acc, false, false);
#pragma unroll
        for (int r = 0; r < 8; ++r) zl[dir][m0 + rbase + r][ccol] = acc[r] + bc;
        __syncthreads();

        // gates: both directions, 2*32*32 cells over 1024 threads
        for (int idx = tid; idx < 2048; idx += 1024) {
            const int d  = idx >> 10;
            const int rc = idx & 1023;
            const int rr = rc >> 5;
            const int u  = rc & 31;
            const float zi = zl[d][rr][u];
            const float zf = zl[d][rr][u + 32];
            const float zg = zl[d][rr][u + 64];
            const float zo = zl[d][rr][u + 96];
            const float ig = sigm(zi), fg = sigm(zf), og = sigm(zo);
            const float cn = fg * cst[d][rr][u] + ig * tanhf(zg);
            cst[d][rr][u] = cn;
            const float hv = og * tanhf(cn);
            hst[d][rr][u] = hv;
            // fragment-order f16 mirror for next step's A(ks=1)
            hsw[d][rr >> 4][slot_lane(u, rr & 15)][slot_j(u)] = (_Float16)hv;
        }
        __syncthreads();
    }

    // final dense: sigmoid([h_f | h_b] @ w_d + b_d) -> (B, 1)
    if (tid < BB) {
        float accv = bd[0];
#pragma unroll
        for (int u = 0; u < UU; ++u)
            accv += hst[0][tid][u] * wd[u] + hst[1][tid][u] * wd[UU + u];
        out[tid] = sigm(accv);
    }
}

// ---------------------------------------------------------------------------
// Host launcher
// ---------------------------------------------------------------------------
extern "C" void kernel_launch(void* const* d_in, const int* in_sizes, int n_in,
                              void* d_out, int out_size, void* d_ws, size_t ws_size,
                              hipStream_t stream) {
    const float* x     = (const float*)d_in[0];
    const float* kconv = (const float*)d_in[1];
    const float* rconv = (const float*)d_in[2];
    const float* bconv = (const float*)d_in[3];
    const float* kf    = (const float*)d_in[4];
    const float* rf    = (const float*)d_in[5];
    const float* bf    = (const float*)d_in[6];
    const float* kb    = (const float*)d_in[7];
    const float* rb    = (const float*)d_in[8];
    const float* bb2   = (const float*)d_in[9];
    const float* wdns  = (const float*)d_in[10];
    const float* bdns  = (const float*)d_in[11];

    char* ws = (char*)d_ws;
    _Float16* WcS   = (_Float16*)(ws);                 // 16384 halfs = 32768 B
    _Float16* WfS   = (_Float16*)(ws + 32768);         //  8192 halfs = 16384 B
    _Float16* WbS   = (_Float16*)(ws + 49152);         //  8192 halfs = 16384 B
    float*    h0    = (float*)(ws + 65536);            // 8192x32 f32 = 1 MB
    float*    h1    = h0 + (size_t)MROWS * FF;         // 1 MB
    float*    cbuf  = h1 + (size_t)MROWS * FF;         // 1 MB
    _Float16* hseqS = (_Float16*)(ws + 65536 + 3u * MROWS * FF * 4u);  // 512 KB
    const unsigned zero_n = 3u * MROWS * FF;           // h0 | h1 | cbuf contiguous

    setup_kernel<<<1024, 256, 0, stream>>>(kconv, rconv, kf, rf, kb, rb,
                                           WcS, WfS, WbS, h0, zero_n);

    for (int t = 0; t < TT; ++t) {
        const float* hp = (t & 1) ? h1 : h0;
        float*       hn = (t & 1) ? h0 : h1;
        convlstm_step_kernel<<<MROWS / 16, 256, 0, stream>>>(x, WcS, bconv, hp, hn, cbuf, t);
    }
    // t = 63 (odd) wrote h0 -> final ConvLSTM hidden sequence lives in h0

    swz_hseq_kernel<<<(SS * 2 * 32 * 16) / 256, 256, 0, stream>>>(h0, hseqS);

    lstm_bidir_kernel<<<1, 1024, 0, stream>>>(hseqS, WfS, WbS, bf, bb2, wdns, bdns,
                                              (float*)d_out);
}
